// TensorProductMultiLayerPerceptron_15461882266099
// MI455X (gfx1250) — matrix operand order repaired and checked by reading, not compiled, not run on hardware
//
#include <hip/hip_runtime.h>

typedef __attribute__((ext_vector_type(16))) _Float16 v16h;
typedef __attribute__((ext_vector_type(8)))  _Float16 v8h;
typedef __attribute__((ext_vector_type(8)))  float    v8f;

#define E_TOT 100000
#define EMB   64
#define HDIM  128
#define MUL   64
#define OUTW  448
#define WAVES 8
#define BLOCK (WAVES * 32)

__device__ __forceinline__ float silu_f(float x) {
    return x / (1.0f + __expf(-x));
}

// D = A*B + C, f16 inputs, f32 accumulate (v_wmma_f32_16x16x32_f16)
__device__ __forceinline__ v8f wmma_f16(v16h a, v16h b, v8f c) {
    return __builtin_amdgcn_wmma_f32_16x16x32_f16(
        /*neg_a=*/false, a, /*neg_b=*/false, b,
        /*c_mod=*/(short)0, c, /*reuse_a=*/false, /*reuse_b=*/false);
}

// B fragment (32x16 f16): lane holds column N = lane%16; kbase = base + 16*lhi;
// halves 0..7 = K base + 8*lhi + (0..7), halves 8..15 = K base + 16 + 8*lhi + (0..7).
__device__ __forceinline__ v16h load_b_lds(const _Float16* col, int kbase) {
    const v8h* p = (const v8h*)(col + kbase - ((kbase & 16) >> 1));
    v8h lo = p[0];
    v8h hi = p[2];
    v16h b;
#pragma unroll
    for (int i = 0; i < 8; ++i) { b[i] = lo[i]; b[8 + i] = hi[i]; }
    return b;
}

// A fragment (16x32 f16) from an LDS row of f16 (row = this lane's M):
// halves 0..7  = K kbase + 8*lhi + (0..7)
// halves 8..15 = K kbase + 16 + 8*lhi + (0..7)
__device__ __forceinline__ v16h load_a_lds(const _Float16* rowp, int kbase, int lhi) {
    const v8h* p0 = (const v8h*)(rowp + kbase + lhi * 8);
    const v8h* p1 = (const v8h*)(rowp + kbase + 16 + lhi * 8);
    v8h lo = *p0;
    v8h hi = *p1;
    v16h a;
#pragma unroll
    for (int i = 0; i < 8; ++i) { a[i] = lo[i]; a[8 + i] = hi[i]; }
    return a;
}

// A fragment from a global f32 row (convert to f16 on the fly)
__device__ __forceinline__ v16h load_a_global_f32(const float* rowp, int kbase, int lhi) {
    const float* q0 = rowp + kbase + lhi * 8;
    const float* q1 = q0 + 16;
    v16h a;
#pragma unroll
    for (int i = 0; i < 8; ++i) {
        a[i]     = (_Float16)q0[i];
        a[8 + i] = (_Float16)q1[i];
    }
    return a;
}

__global__ void __launch_bounds__(BLOCK)
tpmlp_wmma_kernel(const float* __restrict__ emb, const float* __restrict__ x1,
                  const float* __restrict__ x2,  const float* __restrict__ W0,
                  const float* __restrict__ W1,
                  const float* __restrict__ P1, const float* __restrict__ P2,
                  const float* __restrict__ P3, const float* __restrict__ P4,
                  const float* __restrict__ P5,
                  float* __restrict__ out)
{
    extern __shared__ _Float16 smem[];
    _Float16* sW0T = smem;                       // [128][64]  : col-major W0 (n*64 + k)
    _Float16* sW1T = sW0T + HDIM * EMB;          // [128][128] : col-major W1 (n*128 + k)
    _Float16* sPT  = sW1T + HDIM * HDIM;         // [320][128] : col c = p*64+u, (c*128 + k)
    _Float16* sH   = sPT  + 5 * MUL * HDIM;      // per-wave h scratch [16][128]

    const int tid = threadIdx.x;

    // ---- cooperative weight staging: f32 -> f16, transposed for column loads ----
    for (int i = tid; i < HDIM * EMB; i += BLOCK) {
        int n = i / EMB, k = i - n * EMB;
        sW0T[i] = (_Float16)W0[k * HDIM + n];
    }
    for (int i = tid; i < HDIM * HDIM; i += BLOCK) {
        int n = i >> 7, k = i & 127;
        sW1T[i] = (_Float16)W1[k * HDIM + n];
    }
    const float* Ps[5] = {P1, P2, P3, P4, P5};
    for (int i = tid; i < 5 * MUL * HDIM; i += BLOCK) {
        int c = i >> 7, k = i & 127;
        int p = c >> 6, u = c & 63;
        sPT[i] = (_Float16)Ps[p][k * MUL + u];
    }
    __syncthreads();

    const int wave = tid >> 5;
    const int lane = tid & 31;
    const int lh   = lane & 15;   // N (or M) index within 16
    const int lhi  = lane >> 4;   // lane half selector
    const int tile = blockIdx.x * WAVES + wave;
    if (tile * 16 >= E_TOT) return;
    const int row0 = tile * 16;

    _Float16* sHw = sH + wave * (16 * HDIM);

    // prefetch this tile's streaming x1 rows toward L2 (global_prefetch_b8)
    __builtin_prefetch(x1 + (size_t)(row0 + lane) * (4 * MUL), 0, 1);

    // ============ GEMM1: h = silu(emb @ W0 / sqrt(64)) ============
    const float* embrow = emb + (size_t)(row0 + lh) * EMB;
    v16h ae0 = load_a_global_f32(embrow, 0, lhi);
    v16h ae1 = load_a_global_f32(embrow, 32, lhi);
    const float s0 = 0.125f;                 // 1/sqrt(64)
#pragma unroll
    for (int j = 0; j < 8; ++j) {
        v8f acc = {};
        const _Float16* col = sW0T + (j * 16 + lh) * EMB;
        acc = wmma_f16(ae0, load_b_lds(col, 0 + lhi * 16), acc);
        acc = wmma_f16(ae1, load_b_lds(col, 32 + lhi * 16), acc);
#pragma unroll
        for (int v = 0; v < 8; ++v) {
            // C layout: lane holds N = j*16+lh, rows M = v + 8*lhi
            sHw[(v + 8 * lhi) * HDIM + (j * 16 + lh)] = (_Float16)silu_f(acc[v] * s0);
        }
    }
    asm volatile("s_wait_dscnt 0x0" ::: "memory");   // wave-local LDS RAW fence

    // ============ GEMM2: h = silu(h @ W1 / sqrt(128)) ============
    v16h ah[4];
    const _Float16* hrow = sHw + lh * HDIM;
#pragma unroll
    for (int s = 0; s < 4; ++s) ah[s] = load_a_lds(hrow, s * 32, lhi);
    const float s1 = 0.08838834764831845f;   // 1/sqrt(128)
#pragma unroll
    for (int j = 0; j < 8; ++j) {
        v8f acc = {};
        const _Float16* col = sW1T + (j * 16 + lh) * HDIM;
#pragma unroll
        for (int s = 0; s < 4; ++s)
            acc = wmma_f16(ah[s], load_b_lds(col, s * 32 + lhi * 16), acc);
#pragma unroll
        for (int v = 0; v < 8; ++v)
            sHw[(v + 8 * lhi) * HDIM + (j * 16 + lh)] = (_Float16)silu_f(acc[v] * s1);
    }
    asm volatile("s_wait_dscnt 0x0" ::: "memory");

    // reload A fragments of h2 for the projection GEMMs
#pragma unroll
    for (int s = 0; s < 4; ++s) ah[s] = load_a_lds(hrow, s * 32, lhi);

    // ============ GEMM3 + fused tensor product ============
    // per-lane rows: m_v = v + 8*lhi  (matches WMMA C layout)
    float x20[8], x21x[8], x21y[8], x21z[8];
#pragma unroll
    for (int v = 0; v < 8; ++v) {
        size_t e = (size_t)(row0 + v + 8 * lhi);
        const float* q = x2 + e * 4;
        x20[v] = q[0]; x21x[v] = q[1]; x21y[v] = q[2]; x21z[v] = q[3];
    }

    const float inv       = 0.08838834764831845f;  // 1/sqrt(128)
    const float INV_SQRT2 = 0.7071067811865476f;
    const float INV_SQRT3 = 0.5773502691896258f;
    const float INV_SQRT6 = 0.4082482904638631f;

    for (int ub = 0; ub < 4; ++ub) {
        const int u = ub * 16 + lh;    // this lane's channel (0..63)

        // per-row x1 data for this channel
        float x10[8], xa[8], xb[8], xc[8];
#pragma unroll
        for (int v = 0; v < 8; ++v) {
            size_t e = (size_t)(row0 + v + 8 * lhi);
            const float* r = x1 + e * (4 * MUL);
            x10[v] = r[u];
            const float* r1 = r + MUL + 3 * u;
            xa[v] = r1[0]; xb[v] = r1[1]; xc[v] = r1[2];
        }

        auto proj = [&](int p) -> v8f {
            v8f acc = {};
            const _Float16* col = sPT + (size_t)(p * MUL + u) * HDIM;
#pragma unroll
            for (int s = 0; s < 4; ++s)
                acc = wmma_f16(ah[s], load_b_lds(col, s * 32 + lhi * 16), acc);
            return acc;
        };

        // ---- out_0e = INV_SQRT2*(w1*x1_0*x2_0 + w4*dot(x1_1,x2_1)/sqrt(3)) ----
        {
            v8f w1a = proj(0);
            v8f w4a = proj(3);
#pragma unroll
            for (int v = 0; v < 8; ++v) {
                size_t e = (size_t)(row0 + v + 8 * lhi);
                float w1v = w1a[v] * inv, w4v = w4a[v] * inv;
                float dot = xa[v] * x21x[v] + xb[v] * x21y[v] + xc[v] * x21z[v];
                out[e * OUTW + u] =
                    INV_SQRT2 * (w1v * x10[v] * x20[v] + w4v * dot * INV_SQRT3);
            }
        }
        // ---- out_1o = INV_SQRT2*(w2*x1_1*x2_0 + w3*x1_0*x2_1) ----
        {
            v8f w2a = proj(1);
            v8f w3a = proj(2);
#pragma unroll
            for (int v = 0; v < 8; ++v) {
                size_t e = (size_t)(row0 + v + 8 * lhi);
                float w2v = w2a[v] * inv, w3v = w3a[v] * inv;
                float* o = out + e * OUTW + MUL + 3 * u;
                o[0] = INV_SQRT2 * (w2v * xa[v] * x20[v] + w3v * x10[v] * x21x[v]);
                o[1] = INV_SQRT2 * (w2v * xb[v] * x20[v] + w3v * x10[v] * x21y[v]);
                o[2] = INV_SQRT2 * (w2v * xc[v] * x20[v] + w3v * x10[v] * x21z[v]);
            }
        }
        // ---- out_1e = w5 * cross(x1_1, x2_1) / sqrt(6) ----
        {
            v8f w5a = proj(4);
#pragma unroll
            for (int v = 0; v < 8; ++v) {
                size_t e = (size_t)(row0 + v + 8 * lhi);
                float w5v = w5a[v] * inv * INV_SQRT6;
                float cx = xb[v] * x21z[v] - xc[v] * x21y[v];
                float cy = xc[v] * x21x[v] - xa[v] * x21z[v];
                float cz = xa[v] * x21y[v] - xb[v] * x21x[v];
                float* o = out + e * OUTW + MUL + 3 * MUL + 3 * u;
                o[0] = w5v * cx;
                o[1] = w5v * cy;
                o[2] = w5v * cz;
            }
        }
    }
}

extern "C" void kernel_launch(void* const* d_in, const int* in_sizes, int n_in,
                              void* d_out, int out_size, void* d_ws, size_t ws_size,
                              hipStream_t stream) {
    (void)in_sizes; (void)n_in; (void)d_ws; (void)ws_size; (void)out_size;
    const float* emb = (const float*)d_in[0];
    const float* x1  = (const float*)d_in[1];
    const float* x2  = (const float*)d_in[2];
    const float* W0  = (const float*)d_in[3];
    const float* W1  = (const float*)d_in[4];
    const float* P1  = (const float*)d_in[5];
    const float* P2  = (const float*)d_in[6];
    const float* P3  = (const float*)d_in[7];
    const float* P4  = (const float*)d_in[8];
    const float* P5  = (const float*)d_in[9];
    float* out = (float*)d_out;

    const int tiles  = (E_TOT + 15) / 16;            // 6250 row tiles
    const int blocks = (tiles + WAVES - 1) / WAVES;  // 782 blocks
    const size_t smem =
        (size_t)(HDIM * EMB + HDIM * HDIM + 5 * MUL * HDIM + WAVES * 16 * HDIM) *
        sizeof(_Float16);                            // 160 KB

    (void)hipFuncSetAttribute((const void*)tpmlp_wmma_kernel,
                              hipFuncAttributeMaxDynamicSharedMemorySize, (int)smem);

    tpmlp_wmma_kernel<<<blocks, BLOCK, smem, stream>>>(emb, x1, x2, W0, W1,
                                                       P1, P2, P3, P4, P5, out);
}
